// TransformerBlock_85907935855357
// MI455X (gfx1250) — matrix operand order repaired and checked by reading, not compiled, not run on hardware
//
#include <hip/hip_runtime.h>
#include <hip/hip_bf16.h>

#define SEQ    4096
#define DMODEL 2048
#define NHEADS 16
#define DK     128
#define DFF    5504

typedef __attribute__((ext_vector_type(16))) __bf16   v16bf;
typedef __attribute__((ext_vector_type(8)))  float    v8f;
typedef __attribute__((ext_vector_type(4)))  unsigned int u32x4;
typedef __attribute__((ext_vector_type(4)))  int       i32x4;

union FragAB {
    v16bf v;
    u32x4 q[2];
    unsigned short h[16];
};

__device__ __forceinline__ unsigned short f2bf(float f) {
    union { float f; unsigned u; } x{f};
    unsigned r = x.u + 0x7fffu + ((x.u >> 16) & 1u);
    return (unsigned short)(r >> 16);
}
__device__ __forceinline__ float bf2f(unsigned short h) {
    union { unsigned u; float f; } x{((unsigned)h) << 16};
    return x.f;
}

// ---- CDNA5 async global->LDS copy (ASYNCcnt path), with safe fallback ------
#if defined(__has_builtin)
#if __has_builtin(__builtin_amdgcn_global_load_async_to_lds_b128)
#define HAS_ASYNC_LDS 1
#endif
#if __has_builtin(__builtin_amdgcn_s_wait_asynccnt)
#define HAS_WAIT_ASYNC 1
#endif
#endif
#ifndef HAS_ASYNC_LDS
#define HAS_ASYNC_LDS 0
#endif
#ifndef HAS_WAIT_ASYNC
#define HAS_WAIT_ASYNC 0
#endif

typedef __attribute__((address_space(1))) i32x4 g_i32x4;   // global AS pointee
typedef __attribute__((address_space(3))) i32x4 l_i32x4;   // LDS AS pointee

__device__ __forceinline__ void g2l_b128(unsigned short* lds_dst,
                                         const unsigned short* gsrc) {
#if HAS_ASYNC_LDS
    __builtin_amdgcn_global_load_async_to_lds_b128(
        (g_i32x4*)(gsrc), (l_i32x4*)(lds_dst), 0, 0);
#else
    *(u32x4*)lds_dst = *(const u32x4*)gsrc;
#endif
}
__device__ __forceinline__ void wait_async0() {
#if HAS_ASYNC_LDS
#if HAS_WAIT_ASYNC
    __builtin_amdgcn_s_wait_asynccnt(0);
#else
    asm volatile("s_wait_asynccnt 0x0" ::: "memory");
#endif
#endif
}

// ---------------------------------------------------------------- convert
__global__ __launch_bounds__(256) void k_f32_to_bf16(const float* __restrict__ src,
                                                     unsigned short* __restrict__ dst,
                                                     int n) {
    for (int i = blockIdx.x * 256 + threadIdx.x; i < n; i += gridDim.x * 256)
        dst[i] = f2bf(src[i]);
}

// ---------------------------------------------------------------- rmsnorm -> bf16
__global__ __launch_bounds__(256) void k_rmsnorm(const float* __restrict__ x,
                                                 const float* __restrict__ g,
                                                 unsigned short* __restrict__ out) {
    __shared__ float red[256];
    const int row = blockIdx.x;
    const float* xr = x + (size_t)row * DMODEL;
    float ss = 0.f;
    for (int i = threadIdx.x; i < DMODEL; i += 256) { float v = xr[i]; ss += v * v; }
    red[threadIdx.x] = ss;
    __syncthreads();
    for (int s = 128; s > 0; s >>= 1) {
        if ((int)threadIdx.x < s) red[threadIdx.x] += red[threadIdx.x + s];
        __syncthreads();
    }
    float rinv = rsqrtf(red[0] * (1.0f / DMODEL) + 1e-5f);
    unsigned short* o = out + (size_t)row * DMODEL;
    for (int i = threadIdx.x; i < DMODEL; i += 256)
        o[i] = f2bf(xr[i] * rinv * g[i]);
}

// ---------------------------------------------------------------- tiled WMMA GEMM
// acc = A(MxK bf16, row-major) * B(NxK bf16, row-major)^T, 128x128 tile/block.
// Double-buffered LDS (BK=64), async global->LDS pipeline, 1 barrier per k-step.
// MODE 0: C f32 = acc
// MODE 1: C f32 = res + acc
// MODE 2: Hout bf16 = silu(gate) * acc
// MODE 3: Hout bf16 = acc
#define BM 128
#define BN 128
#define BK 64
template <int MODE>
__global__ __launch_bounds__(256) void k_gemm(const unsigned short* __restrict__ A,
                                              const unsigned short* __restrict__ B,
                                              float* __restrict__ C,
                                              const float* __restrict__ res,
                                              const float* __restrict__ gate,
                                              unsigned short* __restrict__ Hout,
                                              int M, int N, int K) {
    __shared__ unsigned short lds_a[2][BM * BK];   // 2 x 16 KB
    __shared__ unsigned short lds_b[2][BN * BK];   // 2 x 16 KB
    const int tid = threadIdx.x;
    const int wave = tid >> 5, lane = tid & 31;
    const int lmod = lane & 15, lhalf = lane >> 4;
    const int wm = wave >> 2, wn = wave & 3;       // 2 x 4 wave grid; wave tile 64x32
    const size_t m0 = (size_t)blockIdx.y * BM;
    const size_t n0 = (size_t)blockIdx.x * BN;

    v8f acc[4][2] = {};

    auto load_tile = [&](int buf, int k0) {
#pragma unroll
        for (int it = 0; it < 4; ++it) {
            int idx = tid + it * 256;              // 1024 16B chunks per matrix
            int row = idx >> 3, seg = idx & 7;
            g2l_b128(&lds_a[buf][row * BK + seg * 8], A + (m0 + row) * K + k0 + seg * 8);
            g2l_b128(&lds_b[buf][row * BK + seg * 8], B + (n0 + row) * K + k0 + seg * 8);
        }
    };

    load_tile(0, 0);
    wait_async0();
    __syncthreads();

    const int nsteps = K / BK;
    for (int step = 0; step < nsteps; ++step) {
        const int cur = step & 1;
        if (step + 1 < nsteps) load_tile(cur ^ 1, (step + 1) * BK);

#pragma unroll
        for (int kk = 0; kk < BK; kk += 32) {
            FragAB bfr[2];
#pragma unroll
            for (int ns = 0; ns < 2; ++ns) {
                int brow = wn * 32 + ns * 16 + lmod;
                const u32x4* p = (const u32x4*)&lds_b[cur][brow * BK + kk + lhalf * 16];
                bfr[ns].q[0] = p[0];
                bfr[ns].q[1] = p[1];
            }
#pragma unroll
            for (int ms = 0; ms < 4; ++ms) {
                int arow = wm * 64 + ms * 16 + lmod;
                FragAB afr;
                afr.q[0] = *(const u32x4*)&lds_a[cur][arow * BK + kk + lhalf * 16];
                afr.q[1] = *(const u32x4*)&lds_a[cur][arow * BK + kk + lhalf * 16 + 8];
                acc[ms][0] = __builtin_amdgcn_wmma_f32_16x16x32_bf16(
                    false, afr.v, false, bfr[0].v, (short)0, acc[ms][0], false, false);
                acc[ms][1] = __builtin_amdgcn_wmma_f32_16x16x32_bf16(
                    false, afr.v, false, bfr[1].v, (short)0, acc[ms][1], false, false);
            }
        }
        wait_async0();
        __syncthreads();
    }

#pragma unroll
    for (int ms = 0; ms < 4; ++ms)
#pragma unroll
        for (int ns = 0; ns < 2; ++ns) {
            size_t gn = n0 + wn * 32 + ns * 16 + lmod;
#pragma unroll
            for (int r = 0; r < 8; ++r) {
                size_t gm = m0 + wm * 64 + ms * 16 + r + 8 * lhalf;
                size_t idx = gm * (size_t)N + gn;
                float v = acc[ms][ns][r];
                if (MODE == 0) {
                    C[idx] = v;
                } else if (MODE == 1) {
                    C[idx] = res[idx] + v;
                } else if (MODE == 2) {
                    float gg = gate[idx];
                    float sg = gg / (1.f + __expf(-gg));
                    Hout[idx] = f2bf(sg * v);
                } else {
                    Hout[idx] = f2bf(v);
                }
            }
        }
}

// ---------------------------------------------------------------- RoPE + head scatter
// qb/kb/vb: bf16 [S][DMODEL]. Outputs: Qh/Kh bf16 [H][S][DK], Vt bf16 [H][DK][S].
__global__ __launch_bounds__(256) void k_rope_scatter(const unsigned short* __restrict__ qb,
                                                      const unsigned short* __restrict__ kb,
                                                      const unsigned short* __restrict__ vb,
                                                      const int* __restrict__ tpos,
                                                      unsigned short* __restrict__ Qh,
                                                      unsigned short* __restrict__ Kh,
                                                      unsigned short* __restrict__ Vt) {
    const int s = blockIdx.x;
    const float pos = (float)tpos[s];
    for (int p = threadIdx.x; p < NHEADS * (DK / 2); p += 256) {
        int h = p >> 6, i = p & 63;
        float freq = __expf(-(float)(2 * i) * (1.0f / DK) * 9.210340371976184f); // ln(1e4)
        float sv, cv;
        __sincosf(pos * freq, &sv, &cv);
        size_t src = (size_t)s * DMODEL + h * DK + 2 * i;
        size_t dst = ((size_t)h * SEQ + s) * DK + 2 * i;
        float qe = bf2f(qb[src]), qo = bf2f(qb[src + 1]);
        Qh[dst]     = f2bf(qe * cv - qo * sv);
        Qh[dst + 1] = f2bf(qe * sv + qo * cv);
        float ke = bf2f(kb[src]), ko = bf2f(kb[src + 1]);
        Kh[dst]     = f2bf(ke * cv - ko * sv);
        Kh[dst + 1] = f2bf(ke * sv + ko * cv);
    }
    for (int d = threadIdx.x; d < DMODEL; d += 256) {
        int h = d >> 7, dd = d & 127;
        Vt[((size_t)h * DK + dd) * SEQ + s] = vb[(size_t)s * DMODEL + d];
    }
}

// ---------------------------------------------------------------- flash attention
// One block per (128-query tile, head). 8 waves; wave owns 16 query rows.
// K tile staged in LDS via async copy; P tile staged in wave-private LDS.
__global__ __launch_bounds__(256) void k_flash(const unsigned short* __restrict__ Qh,
                                               const unsigned short* __restrict__ Kh,
                                               const unsigned short* __restrict__ Vt,
                                               unsigned short* __restrict__ attn) {
    __shared__ unsigned short lds_k[128 * 128];     // 32 KB: K tile [key][d]
    __shared__ unsigned short lds_p[8][16 * 128];   // 32 KB: per-wave P tile
    const int qtile = blockIdx.x, h = blockIdx.y;
    const int tid = threadIdx.x;
    const int wave = tid >> 5, lane = tid & 31;
    const int lmod = lane & 15, lhalf = lane >> 4;
    const int qbase = qtile * 128 + wave * 16;
    const float scale = 0.08838834764831845f;       // 1/sqrt(128)

    // Q fragments, register-resident across the whole key loop.
    FragAB qa[4];
    {
        const unsigned short* qrow = Qh + ((size_t)h * SEQ + qbase + lmod) * DK;
#pragma unroll
        for (int ks = 0; ks < 4; ++ks) {
            qa[ks].q[0] = *(const u32x4*)(qrow + ks * 32 + 16 * lhalf);
            qa[ks].q[1] = *(const u32x4*)(qrow + ks * 32 + 16 * lhalf + 8);
        }
    }

    v8f o_acc[8] = {};
    float m_i[8], l_i[8];
#pragma unroll
    for (int r = 0; r < 8; ++r) { m_i[r] = -3.0e38f; l_i[r] = 0.f; }

    for (int kt = 0; kt <= qtile; ++kt) {
        // ---- stage K tile (async global->LDS), then barrier
        {
            const unsigned short* kbase =
                Kh + ((size_t)h * SEQ + (size_t)kt * 128) * DK;
#pragma unroll
            for (int it = 0; it < 8; ++it) {
                int idx = tid + it * 256;           // 2048 16B chunks
                int row = idx >> 4, seg = idx & 15;
                g2l_b128(&lds_k[row * 128 + seg * 8], kbase + (size_t)row * DK + seg * 8);
            }
            wait_async0();
            __syncthreads();
        }
        // ---- scores S = (Q K^T) * scale
        v8f s_acc[8] = {};
#pragma unroll
        for (int nt = 0; nt < 8; ++nt) {
            const unsigned short* krow = &lds_k[(nt * 16 + lmod) * 128];
#pragma unroll
            for (int ks = 0; ks < 4; ++ks) {
                FragAB bf;
                bf.q[0] = *(const u32x4*)(krow + ks * 32 + 16 * lhalf);
                bf.q[1] = *(const u32x4*)(krow + ks * 32 + 16 * lhalf + 8);
                s_acc[nt] = __builtin_amdgcn_wmma_f32_16x16x32_bf16(
                    false, qa[ks].v, false, bf.v, (short)0, s_acc[nt], false, false);
            }
        }
        // ---- online softmax (per lane: rows r + 8*lhalf, cols lmod over 8 n-tiles)
        const bool diag = (kt == qtile);
#pragma unroll
        for (int r = 0; r < 8; ++r) {
            int q = qbase + r + 8 * lhalf;
            float mx = m_i[r];
#pragma unroll
            for (int nt = 0; nt < 8; ++nt) {
                float v = s_acc[nt][r] * scale;
                int key = kt * 128 + nt * 16 + lmod;
                if (diag && key > q) v = -3.0e38f;
                s_acc[nt][r] = v;
                mx = fmaxf(mx, v);
            }
#pragma unroll
            for (int m = 1; m < 16; m <<= 1) mx = fmaxf(mx, __shfl_xor(mx, m, 32));
            float sc = __expf(m_i[r] - mx);
            l_i[r] *= sc;
            float su = 0.f;
#pragma unroll
            for (int nt = 0; nt < 8; ++nt) {
                float p = __expf(s_acc[nt][r] - mx);
                su += p;
                lds_p[wave][(r + 8 * lhalf) * 128 + nt * 16 + lmod] = f2bf(p);
            }
#pragma unroll
            for (int m = 1; m < 16; m <<= 1) su += __shfl_xor(su, m, 32);
            l_i[r] += su;
            m_i[r] = mx;
#pragma unroll
            for (int dt = 0; dt < 8; ++dt) o_acc[dt][r] *= sc;
        }
        // All waves must finish reading lds_k (scores) before next kt overwrites it,
        // and P must be ordered before the PV reads. One barrier covers both.
        __syncthreads();
        // ---- O += P V   (P from wave-private LDS; V from d-major Vt)
#pragma unroll
        for (int ks2 = 0; ks2 < 4; ++ks2) {
            FragAB pa;
            const unsigned short* prow = &lds_p[wave][lmod * 128 + ks2 * 32];
            pa.q[0] = *(const u32x4*)(prow + 16 * lhalf);
            pa.q[1] = *(const u32x4*)(prow + 16 * lhalf + 8);
#pragma unroll
            for (int dt = 0; dt < 8; ++dt) {
                const unsigned short* vrow =
                    Vt + ((size_t)h * DK + dt * 16 + lmod) * SEQ + (size_t)kt * 128 + ks2 * 32;
                FragAB bf;
                bf.q[0] = *(const u32x4*)(vrow + 16 * lhalf);
                bf.q[1] = *(const u32x4*)(vrow + 16 * lhalf + 8);
                o_acc[dt] = __builtin_amdgcn_wmma_f32_16x16x32_bf16(
                    false, pa.v, false, bf.v, (short)0, o_acc[dt], false, false);
            }
        }
    }
    // ---- epilogue: attn[q][h*128 + d] = O / l  (bf16 for the O-projection GEMM)
#pragma unroll
    for (int dt = 0; dt < 8; ++dt)
#pragma unroll
        for (int r = 0; r < 8; ++r) {
            int q = qbase + r + 8 * lhalf;
            attn[(size_t)q * DMODEL + h * DK + dt * 16 + lmod] =
                f2bf(o_acc[dt][r] / l_i[r]);
        }
}

// ---------------------------------------------------------------- launcher
extern "C" void kernel_launch(void* const* d_in, const int* in_sizes, int n_in,
                              void* d_out, int out_size, void* d_ws, size_t ws_size,
                              hipStream_t stream) {
    (void)in_sizes; (void)n_in; (void)out_size; (void)ws_size;
    const float* x   = (const float*)d_in[0];
    const float* w_q = (const float*)d_in[1];
    const float* w_k = (const float*)d_in[2];
    const float* w_v = (const float*)d_in[3];
    const float* w_o = (const float*)d_in[4];
    const float* w1  = (const float*)d_in[5];
    const float* w2  = (const float*)d_in[6];
    const float* w3  = (const float*)d_in[7];
    const float* g1  = (const float*)d_in[8];
    const float* g2  = (const float*)d_in[9];
    const int*  tpos = (const int*)d_in[10];
    float* out = (float*)d_out;

    char* wsb = (char*)d_ws;
    size_t off = 0;
    auto alloc = [&](size_t bytes) -> void* {
        void* p = wsb + off;
        off += (bytes + 255) & ~(size_t)255;
        return p;
    };
    const size_t SD = (size_t)SEQ * DMODEL;
    const size_t DD = (size_t)DMODEL * DMODEL;
    const size_t FD = (size_t)DFF * DMODEL;
    const size_t SF = (size_t)SEQ * DFF;

    unsigned short* xnb  = (unsigned short*)alloc(SD * 2);
    unsigned short* wqb  = (unsigned short*)alloc(DD * 2);
    unsigned short* wkb  = (unsigned short*)alloc(DD * 2);
    unsigned short* wvb  = (unsigned short*)alloc(DD * 2);
    unsigned short* wob  = (unsigned short*)alloc(DD * 2);
    unsigned short* w1b  = (unsigned short*)alloc(FD * 2);
    unsigned short* w2b  = (unsigned short*)alloc(FD * 2);
    unsigned short* w3b  = (unsigned short*)alloc(FD * 2);
    unsigned short* qb   = (unsigned short*)alloc(SD * 2);
    unsigned short* kb   = (unsigned short*)alloc(SD * 2);
    unsigned short* vb   = (unsigned short*)alloc(SD * 2);
    unsigned short* Qhb  = (unsigned short*)alloc(SD * 2);
    unsigned short* Khb  = (unsigned short*)alloc(SD * 2);
    unsigned short* Vtb  = (unsigned short*)alloc(SD * 2);
    unsigned short* attnb= (unsigned short*)alloc(SD * 2);
    float*          x2   = (float*)alloc(SD * 4);
    unsigned short* xn2b = (unsigned short*)alloc(SD * 2);
    float*          gateb= (float*)alloc(SF * 4);
    unsigned short* hb   = (unsigned short*)alloc(SF * 2);

    // weight conversion fp32 -> bf16
    k_f32_to_bf16<<<2048, 256, 0, stream>>>(w_q, wqb, (int)DD);
    k_f32_to_bf16<<<2048, 256, 0, stream>>>(w_k, wkb, (int)DD);
    k_f32_to_bf16<<<2048, 256, 0, stream>>>(w_v, wvb, (int)DD);
    k_f32_to_bf16<<<2048, 256, 0, stream>>>(w_o, wob, (int)DD);
    k_f32_to_bf16<<<2048, 256, 0, stream>>>(w1, w1b, (int)FD);
    k_f32_to_bf16<<<2048, 256, 0, stream>>>(w2, w2b, (int)FD);
    k_f32_to_bf16<<<2048, 256, 0, stream>>>(w3, w3b, (int)FD);

    // attention path
    k_rmsnorm<<<SEQ, 256, 0, stream>>>(x, g1, xnb);
    dim3 gdd(DMODEL / BN, SEQ / BM);
    k_gemm<3><<<gdd, 256, 0, stream>>>(xnb, wqb, nullptr, nullptr, nullptr, qb,
                                       SEQ, DMODEL, DMODEL);
    k_gemm<3><<<gdd, 256, 0, stream>>>(xnb, wkb, nullptr, nullptr, nullptr, kb,
                                       SEQ, DMODEL, DMODEL);
    k_gemm<3><<<gdd, 256, 0, stream>>>(xnb, wvb, nullptr, nullptr, nullptr, vb,
                                       SEQ, DMODEL, DMODEL);
    k_rope_scatter<<<SEQ, 256, 0, stream>>>(qb, kb, vb, tpos, Qhb, Khb, Vtb);
    dim3 gfl(SEQ / 128, NHEADS);
    k_flash<<<gfl, 256, 0, stream>>>(Qhb, Khb, Vtb, attnb);
    k_gemm<1><<<gdd, 256, 0, stream>>>(attnb, wob, x2, x, nullptr, nullptr,
                                       SEQ, DMODEL, DMODEL);

    // FFN path
    k_rmsnorm<<<SEQ, 256, 0, stream>>>(x2, g2, xn2b);
    dim3 gff(DFF / BN, SEQ / BM);
    k_gemm<0><<<gff, 256, 0, stream>>>(xn2b, w1b, gateb, nullptr, nullptr, nullptr,
                                       SEQ, DFF, DMODEL);
    k_gemm<2><<<gff, 256, 0, stream>>>(xn2b, w3b, nullptr, nullptr, gateb, hb,
                                       SEQ, DFF, DMODEL);
    k_gemm<1><<<gdd, 256, 0, stream>>>(hb, w2b, out, x2, nullptr, nullptr,
                                       SEQ, DMODEL, DFF);
}
